// SutraV054_34084860461141
// MI455X (gfx1250) — matrix operand order repaired and checked by reading, not compiled, not run on hardware
//
#include <hip/hip_runtime.h>
#include <math.h>
#include <stdint.h>

// ---------------- model constants ----------------
#define K_V 50257
#define K_D 768
#define K_FF 1536
#define K_S 6
#define K_B 4
#define K_T 1024
#define K_M (K_B * K_T)   // 4096 tokens
#define K_STEPS 8
#define K_NSLOT 8

typedef __attribute__((ext_vector_type(16))) __bf16 v16bf;
typedef __attribute__((ext_vector_type(8)))  float  v8f;
typedef __bf16 bf16;

union FragBF { uint4 q[2]; v16bf v; };

#define EPI_NONE 0
#define EPI_SILU 1
#define EPI_SOFTPLUS 2

__device__ __forceinline__ float epi_apply(float v, int epi) {
  if (epi == EPI_SILU)     return v / (1.f + __expf(-v));
  if (epi == EPI_SOFTPLUS) return (v > 20.f) ? v : log1pf(__expf(v));
  return v;
}

__device__ __forceinline__ float wave_sum(float v) {
#pragma unroll
  for (int o = 16; o > 0; o >>= 1) v += __shfl_down(v, o);
  return __shfl(v, 0);
}

// pack two f32 -> packed bf16 dword (RNE via v_cvt_pk_bf16_f32)
__device__ __forceinline__ uint32_t pkbf(float a, float b) {
  union { bf16 h[2]; uint32_t u; } t;
  t.h[0] = (bf16)a; t.h[1] = (bf16)b; return t.u;
}

__device__ __forceinline__ v8f wmma_bf16(const FragBF& a, const FragBF& b, v8f c) {
  return __builtin_amdgcn_wmma_f32_16x16x32_bf16(false, a.v, false, b.v,
                                                 (short)0, c, false, false);
}

// =====================================================================
// Weight swizzle: W(K,N) f32 row-major -> bf16 k-pair packed tiles:
// out dword[((k/32)*Np + n)*16 + (k%32)/2] = {W[k][n], W[k+1][n]}
// Np = N padded to 64; pad columns zero-filled -> each 32x64 tile is a
// 4KB contiguous run, ideal for global_load_async_to_lds_b128.
// =====================================================================
__global__ void cvt_swz_k(const float* __restrict__ W, uint32_t* __restrict__ out,
                          int K, int N, int Np)
{
  size_t i = (size_t)blockIdx.x * blockDim.x + threadIdx.x;
  size_t tot = (size_t)(K >> 5) * Np * 16;
  if (i >= tot) return;
  int kp = (int)(i & 15);
  size_t r = i >> 4;
  int n  = (int)(r % Np);
  int kc = (int)(r / Np);
  uint32_t v = 0;
  if (n < N) {
    int k0 = kc * 32 + kp * 2;
    v = pkbf(W[(size_t)k0 * N + n], W[(size_t)(k0 + 1) * N + n]);
  }
  out[i] = v;
}

// =====================================================================
// WMMA GEMM NN: C(M,N) = epi(A(M,K) @ W + bias)[+addc][*rowscale][+=]
// Double-buffered LDS: async B-tile DMA (ASYNCcnt) + A f32->bf16 pack
// for chunk k+1 overlap the WMMAs of chunk k; ONE barrier per chunk.
// Tile 64x64, BK=32, 8 waves -> 2 WMMAs per wave per k-chunk.
// =====================================================================
__global__ __launch_bounds__(256)
void gemm_nn(const float* __restrict__ A, int lda,
             const uint32_t* __restrict__ Bw, int Np,
             const float* __restrict__ bias,
             float* __restrict__ C, int ldc,
             int Mrows, int N, int K,
             int epi, float alpha, float addc,
             const float* __restrict__ rowscale, int rs_stride,
             int accum)
{
  __shared__ __align__(16) bf16     As[2][64][40];   // 80B rows (bank spread)
  __shared__ __align__(16) uint32_t Bsw[2][64][20];  // [n][kpair], 80B rows
  const int tid  = threadIdx.x;
  const int wave = tid >> 5, lane = tid & 31;
  const int bm = blockIdx.y * 64, bn = blockIdx.x * 64;
  const int wm = (wave >> 1) * 16;
  const int wn = (wave & 1) * 32;
  v8f acc0 = {}; v8f acc1 = {};
  const int arow  = tid >> 2;          // 0..63
  const int acol0 = (tid & 3) * 8;     // 0,8,16,24
  const bool avalid = (bm + arow) < Mrows;
  const float* arowp = A + (size_t)(bm + arow) * lda + acol0;
  const uint32_t ldsBlane = (uint32_t)((tid >> 2) * 80 + (tid & 3) * 16);
  const uint32_t ldsB0 = (uint32_t)(uintptr_t)(void*)&Bsw[0][0][0] + ldsBlane;
  const uint32_t ldsB1 = (uint32_t)(uintptr_t)(void*)&Bsw[1][0][0] + ldsBlane;

  if (!avalid) {  // zero invalid A rows once (both buffers), hoisted out of loop
    uint4 z = make_uint4(0, 0, 0, 0);
    *(uint4*)&As[0][arow][acol0] = z;
    *(uint4*)&As[1][arow][acol0] = z;
  }

  auto issue_tile = [&](int k0, int p) {
    // 4KB contiguous pre-swizzled B tile -> LDS via async DMA
    const uint32_t* gsrc = Bw + ((size_t)(k0 >> 5) * Np + bn) * 16 + tid * 4;
    uint32_t dst = p ? ldsB1 : ldsB0;
    asm volatile("global_load_async_to_lds_b128 %0, %1, off"
                 :: "v"(dst), "v"(gsrc) : "memory");
    if (avalid) {  // A tile f32 -> packed bf16
      const float4* ap = (const float4*)(arowp + k0);
      float4 f0 = ap[0], f1 = ap[1];
      uint4 w;
      w.x = pkbf(f0.x, f0.y); w.y = pkbf(f0.z, f0.w);
      w.z = pkbf(f1.x, f1.y); w.w = pkbf(f1.z, f1.w);
      *(uint4*)&As[p][arow][acol0] = w;
    }
  };

  issue_tile(0, 0);
  asm volatile("s_wait_asynccnt 0x0" ::: "memory");
  __syncthreads();

  int p = 0;
  for (int k0 = 0; k0 < K; k0 += 32) {
    const int kn = k0 + 32;
    if (kn < K) issue_tile(kn, p ^ 1);   // prefetch next chunk into alt buffer
    FragBF af, bf0, bf1;
    {
      const int m  = wm + (lane & 15);
      const int kh = lane >> 4;                        // K-half select
      const uint4* rowp = (const uint4*)&As[p][m][0];
      af.q[0] = rowp[2 * kh];
      af.q[1] = rowp[2 * kh + 1];
      const int c0 = wn + (lane & 15);
      const int kq = (lane >> 4) * 2;
      const uint4* cp0 = (const uint4*)&Bsw[p][c0][0];
      bf0.q[0] = cp0[kq]; bf0.q[1] = cp0[kq + 1];
      const uint4* cp1 = (const uint4*)&Bsw[p][c0 + 16][0];
      bf1.q[0] = cp1[kq]; bf1.q[1] = cp1[kq + 1];
    }
    acc0 = wmma_bf16(af, bf0, acc0);
    acc1 = wmma_bf16(af, bf1, acc1);
    asm volatile("s_wait_asynccnt 0x0" ::: "memory");
    __syncthreads();
    p ^= 1;
  }

  const int rh = (lane >> 4) * 8;
  const int cm = lane & 15;
#pragma unroll
  for (int r = 0; r < 8; ++r) {
    const int gm = bm + wm + rh + r;
    if (gm >= Mrows) continue;
#pragma unroll
    for (int half = 0; half < 2; ++half) {
      const int gn = bn + wn + half * 16 + cm;
      if (gn >= N) continue;
      float v = (half ? acc1[r] : acc0[r]) * alpha;
      if (bias) v += bias[gn];
      v = epi_apply(v, epi) + addc;
      if (rowscale) v *= rowscale[(size_t)gm * rs_stride];
      float* cp = C + (size_t)gm * ldc + gn;
      if (accum) *cp += v; else *cp = v;
    }
  }
}

// =====================================================================
// Batched WMMA GEMM NT: C[z] = alpha * A[z](Ma,K) @ B[z](Nb,K)^T
// Attention scores (z=B) and logits final@emb^T (z=1). Double-buffered
// like gemm_nn; B is f32, packed in registers, one b128 LDS store.
// =====================================================================
__global__ __launch_bounds__(256)
void gemm_nt(const float* __restrict__ A, const float* __restrict__ Bm,
             float* __restrict__ C, int Ma, int Nb, int K, float alpha)
{
  __shared__ __align__(16) bf16     As[2][64][40];
  __shared__ __align__(16) uint32_t Bsw[2][64][20];
  const int z = blockIdx.z;
  const float* Ab = A  + (size_t)z * Ma * K;
  const float* Bb = Bm + (size_t)z * Nb * K;
  float*       Cb = C  + (size_t)z * Ma * Nb;
  const int tid  = threadIdx.x;
  const int wave = tid >> 5, lane = tid & 31;
  const int bm = blockIdx.y * 64, bn = blockIdx.x * 64;
  const int wm = (wave >> 1) * 16;
  const int wn = (wave & 1) * 32;
  v8f acc0 = {}; v8f acc1 = {};
  const int arow  = tid >> 2;
  const int acol0 = (tid & 3) * 8;
  const int nbc   = tid >> 2;        // B-tile column
  const int kq0   = (tid & 3) * 8;   // 8 consecutive k per thread
  const bool avalid = (bm + arow) < Ma;
  const bool bvalid = (bn + nbc) < Nb;
  const float* arowp = Ab + (size_t)(bm + arow) * K + acol0;
  const float* browp = Bb + (size_t)(bn + nbc) * K + kq0;

  if (!avalid) {
    uint4 z4 = make_uint4(0, 0, 0, 0);
    *(uint4*)&As[0][arow][acol0] = z4;
    *(uint4*)&As[1][arow][acol0] = z4;
  }
  if (!bvalid) {
    uint4 z4 = make_uint4(0, 0, 0, 0);
    *(uint4*)&Bsw[0][nbc][kq0 >> 1] = z4;
    *(uint4*)&Bsw[1][nbc][kq0 >> 1] = z4;
  }

  auto issue_tile = [&](int k0, int p) {
    if (avalid) {
      const float4* ap = (const float4*)(arowp + k0);
      float4 f0 = ap[0], f1 = ap[1];
      uint4 w;
      w.x = pkbf(f0.x, f0.y); w.y = pkbf(f0.z, f0.w);
      w.z = pkbf(f1.x, f1.y); w.w = pkbf(f1.z, f1.w);
      *(uint4*)&As[p][arow][acol0] = w;
    }
    if (bvalid) {
      const float4* bp = (const float4*)(browp + k0);
      float4 f0 = bp[0], f1 = bp[1];
      uint4 w;
      w.x = pkbf(f0.x, f0.y); w.y = pkbf(f0.z, f0.w);
      w.z = pkbf(f1.x, f1.y); w.w = pkbf(f1.z, f1.w);
      *(uint4*)&Bsw[p][nbc][kq0 >> 1] = w;   // column-major k-pair packed
    }
  };

  issue_tile(0, 0);
  __syncthreads();

  int p = 0;
  for (int k0 = 0; k0 < K; k0 += 32) {
    const int kn = k0 + 32;
    if (kn < K) issue_tile(kn, p ^ 1);
    FragBF af, bf0, bf1;
    {
      const int m  = wm + (lane & 15);
      const int kh = lane >> 4;
      const uint4* rowp = (const uint4*)&As[p][m][0];
      af.q[0] = rowp[2 * kh];
      af.q[1] = rowp[2 * kh + 1];
      const int c0 = wn + (lane & 15);
      const int kq = (lane >> 4) * 2;
      const uint4* cp0 = (const uint4*)&Bsw[p][c0][0];
      bf0.q[0] = cp0[kq]; bf0.q[1] = cp0[kq + 1];
      const uint4* cp1 = (const uint4*)&Bsw[p][c0 + 16][0];
      bf1.q[0] = cp1[kq]; bf1.q[1] = cp1[kq + 1];
    }
    acc0 = wmma_bf16(af, bf0, acc0);
    acc1 = wmma_bf16(af, bf1, acc1);
    __syncthreads();
    p ^= 1;
  }
  const int rh = (lane >> 4) * 8;
  const int cm = lane & 15;
#pragma unroll
  for (int r = 0; r < 8; ++r) {
    const int gm = bm + wm + rh + r;
    if (gm >= Ma) continue;
#pragma unroll
    for (int half = 0; half < 2; ++half) {
      const int gn = bn + wn + half * 16 + cm;
      if (gn >= Nb) continue;
      Cb[(size_t)gm * Nb + gn] = (half ? acc1[r] : acc0[r]) * alpha;
    }
  }
}

// ---------------- misc kernels ----------------
__global__ void embed_k(const int* __restrict__ x, const float* __restrict__ emb,
                        const float* __restrict__ pos, float* __restrict__ h) {
  size_t i = (size_t)blockIdx.x * blockDim.x + threadIdx.x;
  if (i >= (size_t)K_M * K_D) return;
  int m = (int)(i / K_D), d = (int)(i % K_D);
  int t = m % K_T;
  h[i] = emb[(size_t)x[m] * K_D + d] + pos[(size_t)t * K_D + d];
}

__global__ void pi_init_k(float* pi) {
  int i = blockIdx.x * blockDim.x + threadIdx.x;
  if (i < K_M * K_S) pi[i] = ((i % K_S) == 2) ? 1.f : 0.f;
}

// LayerNorm over D=768; 8 rows/block, one wave per row; optional fused add.
__global__ __launch_bounds__(256)
void layernorm_k(const float* __restrict__ x, const float* __restrict__ g,
                 const float* __restrict__ b, float* __restrict__ out,
                 const float* __restrict__ add, float ascale, int nrows)
{
  const int wave = threadIdx.x >> 5, lane = threadIdx.x & 31;
  const int row = blockIdx.x * 8 + wave;
  if (row >= nrows) return;
  const float* xr = x + (size_t)row * K_D;
  float vals[24]; float s = 0.f;
#pragma unroll
  for (int i = 0; i < 24; ++i) { vals[i] = xr[lane + i * 32]; s += vals[i]; }
  const float mean = wave_sum(s) * (1.f / K_D);
  float v2 = 0.f;
#pragma unroll
  for (int i = 0; i < 24; ++i) { float d = vals[i] - mean; v2 += d * d; }
  const float inv = rsqrtf(wave_sum(v2) * (1.f / K_D) + 1e-5f);
  float* orow = out + (size_t)row * K_D;
#pragma unroll
  for (int i = 0; i < 24; ++i) {
    int d = lane + i * 32;
    float r = (vals[i] - mean) * inv * g[d] + b[d];
    if (add) r += ascale * add[(size_t)row * K_D + d];
    orow[d] = r;
  }
}

__global__ void mean_k(const float* __restrict__ mu, float* __restrict__ mumean) {
  int idx = blockIdx.x * blockDim.x + threadIdx.x;
  if (idx >= K_B * K_D) return;
  int b = idx / K_D, d = idx % K_D;
  float s = 0.f;
  for (int t = 0; t < K_T; ++t) s += mu[((size_t)b * K_T + t) * K_D + d];
  mumean[idx] = s * (1.f / K_T);
}

// tiny gating network -> mode bias per batch (B x 36)
__global__ __launch_bounds__(256)
void gate_k(const float* __restrict__ mumean, const float* __restrict__ gw1,
            const float* __restrict__ gb1, const float* __restrict__ gw2,
            const float* __restrict__ gb2, const float* __restrict__ mb,
            float* __restrict__ mode)
{
  __shared__ float g1[K_B][64];
  __shared__ float ml[K_B][2];
  const int tid = threadIdx.x;
  {
    int b = tid >> 6, j = tid & 63;
    float s = gb1[j];
    for (int d = 0; d < K_D; ++d) s += mumean[b * K_D + d] * gw1[d * 64 + j];
    g1[b][j] = s / (1.f + __expf(-s));
  }
  __syncthreads();
  if (tid < 8) {
    int b = tid >> 1, mi = tid & 1;
    float s = gb2[mi];
    for (int j = 0; j < 64; ++j) s += g1[b][j] * gw2[j * 2 + mi];
    ml[b][mi] = s;
  }
  __syncthreads();
  if (tid < K_B * 36) {
    int b = tid / 36, ij = tid % 36;
    float m0 = ml[b][0], m1 = ml[b][1];
    float mx = fmaxf(m0, m1);
    float e0 = __expf(m0 - mx), e1 = __expf(m1 - mx);
    float inv = 1.f / (e0 + e1);
    mode[b * 36 + ij] = (e0 * inv) * mb[ij] + (e1 * inv) * mb[36 + ij];
  }
}

// K = softmax(base + mode) rows; pi_ev = pi @ K
__global__ void transition_k(const float* __restrict__ base, const float* __restrict__ mode,
                             const float* __restrict__ pi, float* __restrict__ pi_ev)
{
  int m = blockIdx.x * blockDim.x + threadIdx.x;
  if (m >= K_M) return;
  int b = m / K_T;
  float Kmat[36];
  for (int i = 0; i < K_S; ++i) {
    float row[K_S]; float mx = -3e38f;
    for (int j = 0; j < K_S; ++j) {
      row[j] = base[(size_t)m * 36 + i * K_S + j] + mode[b * 36 + i * K_S + j];
      mx = fmaxf(mx, row[j]);
    }
    float s = 0.f;
    for (int j = 0; j < K_S; ++j) { row[j] = __expf(row[j] - mx); s += row[j]; }
    float inv = 1.f / s;
    for (int j = 0; j < K_S; ++j) Kmat[i * K_S + j] = row[j] * inv;
  }
  for (int j = 0; j < K_S; ++j) {
    float s = 0.f;
    for (int i = 0; i < K_S; ++i) s += pi[(size_t)m * K_S + i] * Kmat[i * K_S + j];
    pi_ev[(size_t)m * K_S + j] = s;
  }
}

// pi = top2(normalize(pi_ev * softmax(2*evidence)))
__global__ void pi_update_k(const float* __restrict__ pi_ev, const float* __restrict__ ev,
                            float* __restrict__ pi)
{
  int m = blockIdx.x * blockDim.x + threadIdx.x;
  if (m >= K_M) return;
  float e[K_S]; float mx = -3e38f;
  for (int j = 0; j < K_S; ++j) { e[j] = ev[(size_t)m * K_S + j] * 2.f; mx = fmaxf(mx, e[j]); }
  float s = 0.f;
  for (int j = 0; j < K_S; ++j) { e[j] = __expf(e[j] - mx); s += e[j]; }
  float inv = 1.f / s;
  float pn[K_S]; float tot = 0.f;
  for (int j = 0; j < K_S; ++j) { pn[j] = pi_ev[(size_t)m * K_S + j] * e[j] * inv; tot += pn[j]; }
  float tn = fmaxf(tot, 1e-8f);
  for (int j = 0; j < K_S; ++j) pn[j] /= tn;
  int i1 = 0;
  for (int j = 1; j < K_S; ++j) if (pn[j] > pn[i1]) i1 = j;
  int i2 = (i1 == 0) ? 1 : 0;
  for (int j = 0; j < K_S; ++j) if (j != i1 && pn[j] > pn[i2]) i2 = j;
  float s2 = fmaxf(pn[i1] + pn[i2], 1e-8f);
  for (int j = 0; j < K_S; ++j)
    pi[(size_t)m * K_S + j] = (j == i1 || j == i2) ? pn[j] / s2 : 0.f;
}

// local-window softmax + causal top-8 softmax over precomputed scores; gather V
__global__ __launch_bounds__(32)
void router_attend_k(const float* __restrict__ scores, const float* __restrict__ v,
                     float* __restrict__ out)
{
  __shared__ float cv[256]; __shared__ int ci[256];
  __shared__ float tw[8];   __shared__ int tix[8];
  __shared__ float wl[4];   __shared__ int shj0, shn;
  const int m = blockIdx.x;
  const int b = m / K_T, i = m % K_T;
  const int lane = threadIdx.x;
  const float* srow = scores + ((size_t)b * K_T + i) * K_T;
  float bv[8]; int bi[8];
#pragma unroll
  for (int q = 0; q < 8; ++q) { bv[q] = -3e38f; bi[q] = 0; }
  for (int j = lane; j <= i; j += 32) {
    float s = srow[j];
    if (s > bv[7]) {
      int q = 7;
      while (q > 0 && bv[q - 1] < s) { bv[q] = bv[q - 1]; bi[q] = bi[q - 1]; --q; }
      bv[q] = s; bi[q] = j;
    }
  }
#pragma unroll
  for (int q = 0; q < 8; ++q) { cv[lane * 8 + q] = bv[q]; ci[lane * 8 + q] = bi[q]; }
  __syncthreads();
  if (lane == 0) {
    float tv[8]; int tx[8];
#pragma unroll
    for (int q = 0; q < 8; ++q) { tv[q] = -3e38f; tx[q] = 0; }
    for (int c = 0; c < 256; ++c) {
      float s = cv[c];
      if (s > tv[7]) {
        int q = 7;
        while (q > 0 && tv[q - 1] < s) { tv[q] = tv[q - 1]; tx[q] = tx[q - 1]; --q; }
        tv[q] = s; tx[q] = ci[c];
      }
    }
    float mx = tv[0], sum = 0.f, w[8];
#pragma unroll
    for (int q = 0; q < 8; ++q) { w[q] = __expf(tv[q] - mx); sum += w[q]; }
#pragma unroll
    for (int q = 0; q < 8; ++q) { tw[q] = w[q] / sum; tix[q] = tx[q]; }
    int j0 = i - 3; if (j0 < 0) j0 = 0;
    shj0 = j0; shn = i - j0 + 1;
    float lm = -3e38f;
    for (int j = j0; j <= i; ++j) lm = fmaxf(lm, srow[j]);
    float ls = 0.f;
    for (int j = j0; j <= i; ++j) { float e = __expf(srow[j] - lm); wl[j - j0] = e; ls += e; }
    for (int j = j0; j <= i; ++j) wl[j - j0] /= ls;
  }
  __syncthreads();
  const float* vb = v + (size_t)b * K_T * K_D;
  float* orow = out + (size_t)m * K_D;
  for (int d = lane; d < K_D; d += 32) {
    float acc = 0.f;
    for (int q = 0; q < shn; ++q) acc += wl[q] * vb[(size_t)(shj0 + q) * K_D + d];
#pragma unroll
    for (int q = 0; q < 8; ++q)  acc += tw[q] * vb[(size_t)tix[q] * K_D + d];
    orow[d] = acc;
  }
}

// scratchpad read: attn over 8 slots
__global__ __launch_bounds__(256)
void sp_read_k(const float* __restrict__ qsp, const float* __restrict__ mem,
               float* __restrict__ out)
{
  const int wave = threadIdx.x >> 5, lane = threadIdx.x & 31;
  const int m = blockIdx.x * 8 + wave;
  if (m >= K_M) return;
  const int b = m / K_T;
  const float* q  = qsp + (size_t)m * K_D;
  const float* mb = mem + (size_t)b * K_NSLOT * K_D;
  const float scale = rsqrtf((float)K_D);
  float dots[K_NSLOT];
#pragma unroll
  for (int n = 0; n < K_NSLOT; ++n) {
    float s = 0.f;
    for (int d = lane; d < K_D; d += 32) s += q[d] * mb[(size_t)n * K_D + d];
    dots[n] = wave_sum(s) * scale;
  }
  float mx = dots[0];
#pragma unroll
  for (int n = 1; n < K_NSLOT; ++n) mx = fmaxf(mx, dots[n]);
  float sum = 0.f, w[K_NSLOT];
#pragma unroll
  for (int n = 0; n < K_NSLOT; ++n) { w[n] = __expf(dots[n] - mx); sum += w[n]; }
  float inv = 1.f / sum;
  float* orow = out + (size_t)m * K_D;
  for (int d = lane; d < K_D; d += 32) {
    float a = 0.f;
#pragma unroll
    for (int n = 0; n < K_NSLOT; ++n) a += w[n] * inv * mb[(size_t)n * K_D + d];
    orow[d] = a;
  }
}

__global__ void msg_combine_k(float* __restrict__ msg, const float* __restrict__ memctx,
                              const float* __restrict__ pi, const float* __restrict__ pher,
                              float late)
{
  size_t i = (size_t)blockIdx.x * blockDim.x + threadIdx.x;
  if (i >= (size_t)K_M * K_D) return;
  int m = (int)(i / K_D);
  float f = 0.1f;
  if (late > 0.f) f += late * 0.05f / (1.f + __expf(-pher[m]));
  msg[i] += f * memctx[i] * pi[(size_t)m * K_S + 3];
}

__global__ void writer_update_k(const float* __restrict__ muw, float* __restrict__ lam,
                                const float* __restrict__ lam_msg, const float* __restrict__ val,
                                const float* __restrict__ pi, float* __restrict__ mu)
{
  size_t i = (size_t)blockIdx.x * blockDim.x + threadIdx.x;
  if (i >= (size_t)K_M * K_D) return;
  int m = (int)(i / K_D);
  float g  = pi[(size_t)m * K_S + 4];
  float lm = lam_msg[i];
  float l0 = lam[i];
  float ln_ = l0 + g * lm;
  mu[i]  = (l0 * muw[i] + g * lm * val[i]) / ln_;
  lam[i] = ln_;
}

// scratchpad write: one block per (b, slot)
__global__ __launch_bounds__(256)
void sp_write_k(const float* __restrict__ keys, const float* __restrict__ mu,
                const float* __restrict__ spval, const float* __restrict__ pi,
                float* __restrict__ mem)
{
  __shared__ float keysh[K_D];
  __shared__ float lg[K_T];
  __shared__ float red[256];
  const int bn = blockIdx.x;          // 0..B*NSLOT-1
  const int b  = bn / K_NSLOT;
  const int tid = threadIdx.x;
  for (int d = tid; d < K_D; d += 256) keysh[d] = keys[(size_t)bn * K_D + d];
  __syncthreads();
  const float scale = rsqrtf((float)K_D);
  for (int t = tid; t < K_T; t += 256) {
    const float* mrow = mu + ((size_t)b * K_T + t) * K_D;
    float s = 0.f;
    for (int d = 0; d < K_D; ++d) s += keysh[d] * mrow[d];
    float g = pi[((size_t)b * K_T + t) * K_S + 4];
    lg[t] = s * scale + logf(g + 1e-8f);
  }
  __syncthreads();
  float mx = -3e38f;
  for (int t = tid; t < K_T; t += 256) mx = fmaxf(mx, lg[t]);
  red[tid] = mx; __syncthreads();
  for (int o = 128; o > 0; o >>= 1) { if (tid < o) red[tid] = fmaxf(red[tid], red[tid + o]); __syncthreads(); }
  mx = red[0]; __syncthreads();
  float ls = 0.f;
  for (int t = tid; t < K_T; t += 256) { float e = __expf(lg[t] - mx); lg[t] = e; ls += e; }
  red[tid] = ls; __syncthreads();
  for (int o = 128; o > 0; o >>= 1) { if (tid < o) red[tid] += red[tid + o]; __syncthreads(); }
  const float inv = 1.f / red[0];
  __syncthreads();
  for (int d = tid; d < K_D; d += 256) {
    float acc = 0.f;
    for (int t = 0; t < K_T; ++t) acc += lg[t] * spval[((size_t)b * K_T + t) * K_D + d];
    float* mp = mem + (size_t)bn * K_D + d;
    *mp = 0.9f * (*mp) + 0.1f * acc * inv;
  }
}

__global__ __launch_bounds__(256)
void pheromone_k(const float* __restrict__ mu, float* __restrict__ mu_prev,
                 const float* __restrict__ pi, float* __restrict__ pher, float late)
{
  const int wave = threadIdx.x >> 5, lane = threadIdx.x & 31;
  const int m = blockIdx.x * 8 + wave;
  if (m >= K_M) return;
  const float* mr = mu + (size_t)m * K_D;
  float* pr = mu_prev + (size_t)m * K_D;
  float s = 0.f;
#pragma unroll
  for (int i = 0; i < 24; ++i) {
    int d = lane + i * 32;
    float dd = mr[d] - pr[d];
    s += dd * dd;
    pr[d] = mr[d];
  }
  s = wave_sum(s);
  if (lane == 0) {
    float dm = sqrtf(s * (1.f / K_D));
    float dep = pi[(size_t)m * K_S + 4] * tanhf(dm);
    pher[m] = 0.9f * pher[m] + late * dep;
  }
}

// =====================================================================
// host orchestration
// =====================================================================
static inline size_t cdiv(size_t a, size_t b) { return (a + b - 1) / b; }
static inline int nup64(int n) { return (int)((n + 63) / 64) * 64; }

static void gemmNN(hipStream_t s, const float* A, int lda, const uint32_t* Bw, int Np,
                   const float* bias, float* C, int ldc, int M_, int N_, int K_,
                   int epi, float alpha, float addc, const float* rs, int rss, int accum)
{
  dim3 g((unsigned)cdiv(N_, 64), (unsigned)cdiv(M_, 64));
  gemm_nn<<<g, 256, 0, s>>>(A, lda, Bw, Np, bias, C, ldc, M_, N_, K_,
                            epi, alpha, addc, rs, rss, accum);
}

extern "C" void kernel_launch(void* const* d_in, const int* in_sizes, int n_in,
                              void* d_out, int out_size, void* d_ws, size_t ws_size,
                              hipStream_t stream)
{
  // input mapping: {'x': ..., 'params': ...} -> x first; fall back to x-last.
  int base = 1;
  const int* xin;
  if (in_sizes[0] == K_M) { xin = (const int*)d_in[0]; base = 1; }
  else { xin = (const int*)d_in[n_in - 1]; base = 0; }
#define P(i) ((const float*)d_in[base + (i)])
  const float* emb = P(0);  const float* pos = P(1);
  const float *init_mu_w = P(2), *init_mu_b = P(3), *init_lam_w = P(4), *init_lam_b = P(5);
  const float *tr_w1 = P(6), *tr_b1 = P(7), *tr_w2 = P(8), *tr_b2 = P(9);
  const float *gw1 = P(10), *gb1 = P(11), *gw2 = P(12), *gb2 = P(13), *modeb = P(14);
  const float *sb_w1 = P(15), *sb_b1 = P(16), *sb_w2 = P(17), *sb_b2 = P(18);
  const float *ev_w = P(19), *ev_b = P(20);
  const float *rq_w = P(21), *rq_b = P(22), *rk_w = P(23), *rk_b = P(24);
  const float *rv_w = P(25), *rv_b = P(26), *ro_w = P(27), *ro_b = P(28);
  const float *wval_w = P(29), *wval_b = P(30), *wlam_w = P(31), *wlam_b = P(32);
  const float *spr_w = P(33), *spk_w = P(34), *spv_w = P(35);
  const float *g_preb = P(36), *b_preb = P(37), *g_postb = P(38), *b_postb = P(39);
  const float *g_prer = P(40), *b_prer = P(41), *g_postr = P(42), *b_postr = P(43);
  const float *g_prew = P(44), *b_prew = P(45), *g_postw = P(46), *b_postw = P(47);
  const float *g_fin = P(48), *b_fin = P(49);
#undef P

  // workspace carve-out
  char* wp = (char*)d_ws;
  size_t off = 0;
  auto alloc = [&](size_t bytes) -> void* {
    void* p = wp + off; off += (bytes + 255) & ~(size_t)255; return p;
  };
  const size_t MD = (size_t)K_M * K_D;
  const int NP_D  = nup64(K_D);    // 768
  const int NP_FF = nup64(K_FF);   // 1536
  const int NP_256 = 256, NP_36 = 64, NP_S = 64;

  // swizzled bf16 weights (K*Np*2 bytes each)
  auto walloc = [&](int K_, int Np_) { return (uint32_t*)alloc((size_t)K_ * Np_ * 2); };
  uint32_t* wmu  = walloc(K_D, NP_D);
  uint32_t* wlam = walloc(K_D, NP_D);
  uint32_t* wt1  = walloc(K_D, NP_256);
  uint32_t* wt2  = walloc(256, NP_36);
  uint32_t* wsb1 = walloc(K_S * K_D, NP_FF);   // 6 stage blocks, contiguous
  uint32_t* wsb2 = walloc(K_S * K_FF, NP_D);
  uint32_t* wev  = walloc(K_D, NP_S);
  uint32_t* wrq  = walloc(K_D, NP_D);
  uint32_t* wrk  = walloc(K_D, NP_D);
  uint32_t* wrv  = walloc(K_D, NP_D);
  uint32_t* wro  = walloc(K_D, NP_D);
  uint32_t* wwv  = walloc(K_D, NP_D);
  uint32_t* wwl  = walloc(K_D, NP_D);
  uint32_t* wsr  = walloc(K_D, NP_D);
  uint32_t* wsk  = walloc(K_D, NP_D);
  uint32_t* wsv  = walloc(K_D, NP_D);
  // f32 activations
  float* h       = (float*)alloc(MD * 4);
  float* mu      = (float*)alloc(MD * 4);
  float* lam     = (float*)alloc(MD * 4);
  float* muprev  = (float*)alloc(MD * 4);
  float* tmpA    = (float*)alloc(MD * 4);
  float* tmpB    = (float*)alloc(MD * 4);
  float* stageout= (float*)alloc(MD * 4);
  float* stageh  = (float*)alloc((size_t)K_M * K_FF * 4);
  float* qb      = (float*)alloc(MD * 4);
  float* kb      = (float*)alloc(MD * 4);
  float* vb      = (float*)alloc(MD * 4);
  float* msg     = (float*)alloc(MD * 4);
  float* scores  = (float*)alloc((size_t)K_B * K_T * K_T * 4);  // reused as spval
  float* h1      = (float*)alloc((size_t)K_M * 256 * 4);
  float* basep   = (float*)alloc((size_t)K_M * 36 * 4);
  float* pi      = (float*)alloc((size_t)K_M * K_S * 4);
  float* piev    = (float*)alloc((size_t)K_M * K_S * 4);
  float* evid    = (float*)alloc((size_t)K_M * K_S * 4);
  float* mumean  = (float*)alloc((size_t)K_B * K_D * 4);
  float* modeB   = (float*)alloc((size_t)K_B * 36 * 4);
  float* mem     = (float*)alloc((size_t)K_B * K_NSLOT * K_D * 4);
  float* keys    = (float*)alloc((size_t)K_B * K_NSLOT * K_D * 4);
  float* pher    = (float*)alloc((size_t)K_M * 4);
  float* spval   = scores;  // alias: scores dead by sp_write time
  (void)ws_size;

  // weight conversion f32 -> swizzled bf16
  auto swz = [&](const float* src, uint32_t* dst, int K_, int N_, int Np_) {
    size_t tot = (size_t)(K_ >> 5) * Np_ * 16;
    cvt_swz_k<<<(unsigned)cdiv(tot, 256), 256, 0, stream>>>(src, dst, K_, N_, Np_);
  };
  swz(init_mu_w, wmu, K_D, K_D, NP_D);   swz(init_lam_w, wlam, K_D, K_D, NP_D);
  swz(tr_w1, wt1, K_D, 256, NP_256);     swz(tr_w2, wt2, 256, 36, NP_36);
  for (int s = 0; s < K_S; ++s) {
    swz(sb_w1 + (size_t)s * K_D * K_FF, wsb1 + (size_t)s * K_D * NP_FF / 2,
        K_D, K_FF, NP_FF);
    swz(sb_w2 + (size_t)s * K_FF * K_D, wsb2 + (size_t)s * K_FF * NP_D / 2,
        K_FF, K_D, NP_D);
  }
  swz(ev_w, wev, K_D, K_S, NP_S);
  swz(rq_w, wrq, K_D, K_D, NP_D);  swz(rk_w, wrk, K_D, K_D, NP_D);
  swz(rv_w, wrv, K_D, K_D, NP_D);  swz(ro_w, wro, K_D, K_D, NP_D);
  swz(wval_w, wwv, K_D, K_D, NP_D); swz(wlam_w, wwl, K_D, K_D, NP_D);
  swz(spr_w, wsr, K_D, K_D, NP_D);  swz(spk_w, wsk, K_D, K_D, NP_D);
  swz(spv_w, wsv, K_D, K_D, NP_D);

  const float iscale = rsqrtf((float)K_D);
  const unsigned EWB = (unsigned)cdiv(MD, 256);

  // ---- init ----
  embed_k<<<EWB, 256, 0, stream>>>(xin, emb, pos, h);
  gemmNN(stream, h, K_D, wmu, NP_D, init_mu_b, mu, K_D, K_M, K_D, K_D,
         EPI_NONE, 1.f, 0.f, nullptr, 0, 0);
  gemmNN(stream, h, K_D, wlam, NP_D, init_lam_b, lam, K_D, K_M, K_D, K_D,
         EPI_SOFTPLUS, 1.f, 0.1f, nullptr, 0, 0);
  pi_init_k<<<(unsigned)cdiv(K_M * K_S, 256), 256, 0, stream>>>(pi);
  hipMemsetAsync(mem, 0, (size_t)K_B * K_NSLOT * K_D * 4, stream);
  hipMemsetAsync(pher, 0, (size_t)K_M * 4, stream);
  hipMemcpyAsync(muprev, mu, MD * 4, hipMemcpyDeviceToDevice, stream);

  // ---- refinement steps ----
  for (int t = 0; t < K_STEPS; ++t) {
    const float late = (t >= 3) ? 1.f : 0.f;
    // transition kernel
    gemmNN(stream, mu, K_D, wt1, NP_256, tr_b1, h1, 256, K_M, 256, K_D,
           EPI_SILU, 1.f, 0.f, nullptr, 0, 0);
    gemmNN(stream, h1, 256, wt2, NP_36, tr_b2, basep, 36, K_M, 36, 256,
           EPI_NONE, 1.f, 0.f, nullptr, 0, 0);
    mean_k<<<(unsigned)cdiv(K_B * K_D, 256), 256, 0, stream>>>(mu, mumean);
    gate_k<<<1, 256, 0, stream>>>(mumean, gw1, gb1, gw2, gb2, modeb, modeB);
    transition_k<<<(unsigned)cdiv(K_M, 256), 256, 0, stream>>>(basep, modeB, pi, piev);
    // stage bank (uses old pi)
    layernorm_k<<<K_M / 8, 256, 0, stream>>>(mu, g_preb, b_preb, tmpA, nullptr, 0.f, K_M);
    gemmNN(stream, tmpA, K_D, wev, NP_S, ev_b, evid, K_S, K_M, K_S, K_D,
           EPI_NONE, 1.f, 0.f, nullptr, 0, 0);
    for (int s = 0; s < K_S; ++s) {
      gemmNN(stream, tmpA, K_D, wsb1 + (size_t)s * K_D * NP_FF / 2, NP_FF,
             sb_b1 + (size_t)s * K_FF, stageh, K_FF, K_M, K_FF, K_D,
             EPI_SILU, 1.f, 0.f, nullptr, 0, 0);
      gemmNN(stream, stageh, K_FF, wsb2 + (size_t)s * K_FF * NP_D / 2, NP_D,
             sb_b2 + (size_t)s * K_D, stageout, K_D, K_M, K_D, K_FF,
             EPI_NONE, 1.f, 0.f, pi + s, K_S, (s > 0) ? 1 : 0);
    }
    layernorm_k<<<K_M / 8, 256, 0, stream>>>(stageout, g_postb, b_postb, stageout,
                                             nullptr, 0.f, K_M);
    pi_update_k<<<(unsigned)cdiv(K_M, 256), 256, 0, stream>>>(piev, evid, pi);
    // router (new pi)
    layernorm_k<<<K_M / 8, 256, 0, stream>>>(mu, g_prer, b_prer, tmpA, nullptr, 0.f, K_M);
    gemmNN(stream, tmpA, K_D, wrq, NP_D, rq_b, qb, K_D, K_M, K_D, K_D, EPI_NONE, 1.f, 0.f, nullptr, 0, 0);
    gemmNN(stream, tmpA, K_D, wrk, NP_D, rk_b, kb, K_D, K_M, K_D, K_D, EPI_NONE, 1.f, 0.f, nullptr, 0, 0);
    gemmNN(stream, tmpA, K_D, wrv, NP_D, rv_b, vb, K_D, K_M, K_D, K_D, EPI_NONE, 1.f, 0.f, nullptr, 0, 0);
    {
      dim3 g((unsigned)cdiv(K_T, 64), (unsigned)cdiv(K_T, 64), K_B);
      gemm_nt<<<g, 256, 0, stream>>>(qb, kb, scores, K_T, K_T, K_D, iscale);
    }
    router_attend_k<<<K_M, 32, 0, stream>>>(scores, vb, tmpB);
    gemmNN(stream, tmpB, K_D, wro, NP_D, ro_b, msg, K_D, K_M, K_D, K_D,
           EPI_NONE, 1.f, 0.f, pi + 3, K_S, 0);      // * pi[:,3]
    layernorm_k<<<K_M / 8, 256, 0, stream>>>(msg, g_postr, b_postr, msg, nullptr, 0.f, K_M);
    // scratchpad read
    gemmNN(stream, mu, K_D, wsr, NP_D, nullptr, qb, K_D, K_M, K_D, K_D,
           EPI_NONE, 1.f, 0.f, nullptr, 0, 0);
    sp_read_k<<<K_M / 8, 256, 0, stream>>>(qb, mem, vb);
    msg_combine_k<<<EWB, 256, 0, stream>>>(msg, vb, pi, pher, late);
    // bayesian writer
    layernorm_k<<<K_M / 8, 256, 0, stream>>>(mu, g_prew, b_prew, tmpB, nullptr, 0.f, K_M);
    gemmNN(stream, msg, K_D, wwl, NP_D, wlam_b, qb, K_D, K_M, K_D, K_D,
           EPI_SOFTPLUS, 1.f, 0.f, nullptr, 0, 0);
    gemmNN(stream, msg, K_D, wwv, NP_D, wval_b, kb, K_D, K_M, K_D, K_D,
           EPI_NONE, 1.f, 0.f, nullptr, 0, 0);
    writer_update_k<<<EWB, 256, 0, stream>>>(tmpB, lam, qb, kb, pi, mu);
    layernorm_k<<<K_M / 8, 256, 0, stream>>>(mu, g_postw, b_postw, mu, stageout, 0.1f, K_M);
    // scratchpad write
    gemmNN(stream, mem, K_D, wsk, NP_D, nullptr, keys, K_D, K_B * K_NSLOT, K_D, K_D,
           EPI_NONE, 1.f, 0.f, nullptr, 0, 0);
    gemmNN(stream, mu, K_D, wsv, NP_D, nullptr, spval, K_D, K_M, K_D, K_D,
           EPI_NONE, 1.f, 0.f, nullptr, 0, 0);
    sp_write_k<<<K_B * K_NSLOT, 256, 0, stream>>>(keys, mu, spval, pi, mem);
    pheromone_k<<<K_M / 8, 256, 0, stream>>>(mu, muprev, pi, pher, late);
  }

  // ---- head: logits = LN(mu) @ emb^T / sqrt(D) ----
  layernorm_k<<<K_M / 8, 256, 0, stream>>>(mu, g_fin, b_fin, tmpA, nullptr, 0.f, K_M);
  {
    dim3 g((unsigned)cdiv(K_V, 64), (unsigned)cdiv(K_M, 64), 1);
    gemm_nt<<<g, 256, 0, stream>>>(tmpA, emb, (float*)d_out, K_M, K_V, K_D, iscale);
  }
  (void)out_size; (void)n_in;
}
